// SingleHeadAttentionLayer1_22763326669315
// MI455X (gfx1250) — matrix operand order repaired and checked by reading, not compiled, not run on hardware
//
#include <hip/hip_runtime.h>

// ---------------------------------------------------------------------------
// Single-head attention for MI455X (gfx1250, wave32, WMMA).
//   Qp = (q @ Wq^T + bq) * (1/sqrt(64))   -> bf16 [4096][64]
//   Kp =  k @ Wk^T + bk                    -> bf16 [8192][64]
//   Vp =  v @ Wv^T + bv                    -> bf16 [8192][64]
//   out = softmax(Qp Kp^T) @ Vp            (flash / online softmax, f32 acc)
//
// Transposed formulation: S^T = K @ Q^T and O^T = V^T @ P^T, so each lane owns
// one query column: softmax stats are lane-scalars, the row-max needs a single
// cross-half shuffle, and row-sums come from a ones-matrix WMMA.
// K tiles are staged with CDNA5 async global->LDS copies (ASYNCcnt domain).
// ---------------------------------------------------------------------------

typedef __attribute__((ext_vector_type(16))) __bf16        v16bf;
typedef __attribute__((ext_vector_type(8)))  __bf16        v8bf;
typedef __attribute__((ext_vector_type(8)))  float         v8f;
typedef __attribute__((ext_vector_type(8)))  unsigned short v8u16;

#define NQ_  4096
#define NK_  8192
#define AD_  64      // attention (projected) dim
#define DV_  64      // value dim

__device__ __forceinline__ unsigned short f2bf(float f) {
    __bf16 h = (__bf16)f;                       // RNE convert
    return __builtin_bit_cast(unsigned short, h);
}

__device__ __forceinline__ v16bf cat8(v8bf lo, v8bf hi) {
    return __builtin_shufflevector(lo, hi, 0,1,2,3,4,5,6,7,8,9,10,11,12,13,14,15);
}

// ---------------------------------------------------------------------------
// Linear projection: Y[row][col] = (b[col] + sum_i X[row][i]*W[col][i]) * scale
// grid = rows, block = 64 (out_dim), bf16 output.
// ---------------------------------------------------------------------------
__global__ void proj_kernel(const float* __restrict__ X,
                            const float* __restrict__ W,
                            const float* __restrict__ B,
                            unsigned short* __restrict__ Y,
                            int in_dim, float scale) {
    const int row = blockIdx.x;
    const int col = threadIdx.x;                 // 0..63
    const float* xr = X + (size_t)row * in_dim;
    const float* wr = W + (size_t)col * in_dim;
    float acc = B[col];
#pragma unroll 8
    for (int i = 0; i < in_dim; ++i) acc = fmaf(xr[i], wr[i], acc);
    Y[(size_t)row * 64 + col] = f2bf(acc * scale);
}

// ---------------------------------------------------------------------------
// Flash attention kernel. 256 threads = 8 waves; each wave owns 16 query rows.
// ---------------------------------------------------------------------------
__launch_bounds__(256)
__global__ void flash_attn_kernel(const unsigned short* __restrict__ Qp,
                                  const unsigned short* __restrict__ Kp,
                                  const unsigned short* __restrict__ Vp,
                                  float* __restrict__ Out) {
    __shared__ __attribute__((aligned(32))) unsigned short Kt[32 * 64];  // [key][feat]
    __shared__ __attribute__((aligned(32))) unsigned short Vt[64 * 32];  // [col][key] (transposed)
    __shared__ __attribute__((aligned(32))) unsigned short Pw[8][16 * 32]; // [query][key]

    const int tid      = threadIdx.x;
    const int wave     = tid >> 5;
    const int lane     = tid & 31;
    const int l16      = lane & 15;              // query index within wave tile
    const int laneHalf = lane >> 4;              // 0: lanes 0-15, 1: lanes 16-31
    const int kb       = laneHalf * 8;           // A-fragment K sub-offset (elems)
    const int fo       = laneHalf * 16;          // unused (B-fragments also use kb)

    const int qbase = (blockIdx.x * 8 + wave) * 16;

    // LDS byte offset of this thread's K-tile destination (for async copy)
    const unsigned kt_off = (unsigned)(size_t)(const void*)Kt + (unsigned)tid * 16u;

    // ---- Q as B-fragments of S^T = K @ Q^T : B[f][q] = Qp[q][f], table k order (as K)
    const unsigned short* qrow = Qp + (size_t)(qbase + l16) * AD_;
    const v16bf bq0 = cat8(*(const v8bf*)(qrow + kb), *(const v8bf*)(qrow + 16 + kb));        // f 0..31
    const v16bf bq1 = cat8(*(const v8bf*)(qrow + 32 + kb), *(const v8bf*)(qrow + 48 + kb));   // f 32..63

    // all-ones A fragment: lsum = ones @ P^T gives per-query sums
    v16bf aones;
#pragma unroll
    for (int i = 0; i < 16; ++i) aones[i] = (__bf16)1.0f;

    v8f ot0 = {}, ot1 = {}, ot2 = {}, ot3 = {};  // O^T: 64x16 f32 accumulator
    float mrow = -1.0e30f, lrow = 0.0f;          // per-lane (per-query) stats

    for (int kv = 0; kv < NK_; kv += 32) {
        __syncthreads();
        // ---- stage K tile: CDNA5 async global->LDS (no VGPR round-trip).
        //      Tile is contiguous: 32 keys x 64 feats x 2B = 4KB; 16B/lane.
        {
            const unsigned short* gsrc = Kp + (size_t)kv * AD_ + (size_t)tid * 8;
            asm volatile("global_load_async_to_lds_b128 %0, %1, off"
                         :: "v"(kt_off), "v"(gsrc) : "memory");
        }
        // ---- stage V tile transposed: coalesced 16B global load,
        //      scatter-transpose into LDS (Vt[c][k] = Vp[kv+k][c])
        {
            const int key = tid >> 3;            // 0..31
            const int c0  = (tid & 7) * 8;       // 0..56
            v8u16 tv = *(const v8u16*)(Vp + (size_t)(kv + key) * DV_ + c0);
#pragma unroll
            for (int j = 0; j < 8; ++j)
                Vt[(c0 + j) * 32 + key] = tv[j];
        }
        if (kv + 32 < NK_) {                     // global_prefetch_b8 for next tiles
            __builtin_prefetch(Kp + (size_t)(kv + 32) * AD_ + tid * 8, 0, 3);
            __builtin_prefetch(Vp + (size_t)(kv + 32) * DV_ + tid * 8, 0, 3);
        }
        asm volatile("s_wait_asynccnt 0" ::: "memory");  // K tile landed in LDS
        __syncthreads();

        // ---- S^T = K Q^T : two 16x16 f32 tiles (keys 0-15 and 16-31 as M)
        v8f s0 = {}, s1 = {};
        {
            const unsigned short* r0 = Kt + l16 * 64;          // keys kv+0..15
            const unsigned short* r1 = Kt + (l16 + 16) * 64;   // keys kv+16..31
            v16bf aK0a = cat8(*(const v8bf*)(r0 + kb),      *(const v8bf*)(r0 + 16 + kb));
            v16bf aK0b = cat8(*(const v8bf*)(r0 + 32 + kb), *(const v8bf*)(r0 + 48 + kb));
            v16bf aK1a = cat8(*(const v8bf*)(r1 + kb),      *(const v8bf*)(r1 + 16 + kb));
            v16bf aK1b = cat8(*(const v8bf*)(r1 + 32 + kb), *(const v8bf*)(r1 + 48 + kb));
            s0 = __builtin_amdgcn_wmma_f32_16x16x32_bf16(false, aK0a, false, bq0, (short)0, s0, false, false);
            s0 = __builtin_amdgcn_wmma_f32_16x16x32_bf16(false, aK0b, false, bq1, (short)0, s0, false, false);
            s1 = __builtin_amdgcn_wmma_f32_16x16x32_bf16(false, aK1a, false, bq0, (short)0, s1, false, false);
            s1 = __builtin_amdgcn_wmma_f32_16x16x32_bf16(false, aK1b, false, bq1, (short)0, s1, false, false);
        }

        // ---- online softmax: lane holds query l16, keys kv + 8*laneHalf + r
        //      (s0) and kv + 16 + 8*laneHalf + r (s1)
        {
            v8f sm;
#pragma unroll
            for (int r = 0; r < 8; ++r) sm[r] = fmaxf(s0[r], s1[r]);
            float v = fmaxf(fmaxf(fmaxf(sm[0], sm[1]), fmaxf(sm[2], sm[3])),
                            fmaxf(fmaxf(sm[4], sm[5]), fmaxf(sm[6], sm[7])));
            v = fmaxf(v, __shfl_xor(v, 16, 32));       // other half's 16 keys
            const float mnew = fmaxf(mrow, v);
            const float scal = __expf(mrow - mnew);
            mrow = mnew;
            v8u16 p0, p1;
#pragma unroll
            for (int r = 0; r < 8; ++r) {
                p0[r] = f2bf(__expf(s0[r] - mnew));
                p1[r] = f2bf(__expf(s1[r] - mnew));
            }
            // P^T staged as Pw[query][key]: contiguous 8-key runs per lane
            unsigned short* pr = Pw[wave] + l16 * 32 + kb;
            *(v8u16*)(pr)      = p0;                   // keys 8*half + 0..7
            *(v8u16*)(pr + 16) = p1;                   // keys 16 + 8*half + 0..7
            // rescale O^T
#pragma unroll
            for (int r = 0; r < 8; ++r) {
                ot0[r] *= scal; ot1[r] *= scal; ot2[r] *= scal; ot3[r] *= scal;
            }
            lrow *= scal;
        }
        asm volatile("s_wait_dscnt 0" ::: "memory");   // same-wave LDS store->load

        // ---- O^T += V^T @ P^T ; row sums via ones @ P^T
        {
            const v16bf bp = cat8(*(const v8bf*)(Pw[wave] + l16 * 32 + kb), *(const v8bf*)(Pw[wave] + l16 * 32 + 16 + kb));
            v8f lsum = {};
            lsum = __builtin_amdgcn_wmma_f32_16x16x32_bf16(false, aones, false, bp, (short)0, lsum, false, false);
            const unsigned short* v0 = Vt + (0  + l16) * 32;
            const unsigned short* v1 = Vt + (16 + l16) * 32;
            const unsigned short* v2 = Vt + (32 + l16) * 32;
            const unsigned short* v3 = Vt + (48 + l16) * 32;
            v16bf aV0 = cat8(*(const v8bf*)(v0 + kb), *(const v8bf*)(v0 + 16 + kb));
            v16bf aV1 = cat8(*(const v8bf*)(v1 + kb), *(const v8bf*)(v1 + 16 + kb));
            v16bf aV2 = cat8(*(const v8bf*)(v2 + kb), *(const v8bf*)(v2 + 16 + kb));
            v16bf aV3 = cat8(*(const v8bf*)(v3 + kb), *(const v8bf*)(v3 + 16 + kb));
            ot0 = __builtin_amdgcn_wmma_f32_16x16x32_bf16(false, aV0, false, bp, (short)0, ot0, false, false);
            ot1 = __builtin_amdgcn_wmma_f32_16x16x32_bf16(false, aV1, false, bp, (short)0, ot1, false, false);
            ot2 = __builtin_amdgcn_wmma_f32_16x16x32_bf16(false, aV2, false, bp, (short)0, ot2, false, false);
            ot3 = __builtin_amdgcn_wmma_f32_16x16x32_bf16(false, aV3, false, bp, (short)0, ot3, false, false);
            lrow += lsum[0];                           // same value in every row m
        }
    }

    // ---- epilogue: O^T/l -> out[4096][64] f32; lane writes its query's row
    {
        const float inv = 1.0f / lrow;
        float* orow = Out + (size_t)(qbase + l16) * 64 + 8 * laneHalf;
        v8f t;
#pragma unroll
        for (int r = 0; r < 8; ++r) t[r] = ot0[r] * inv;
        *(v8f*)(orow) = t;                              // cols  0..15 (half-split)
#pragma unroll
        for (int r = 0; r < 8; ++r) t[r] = ot1[r] * inv;
        *(v8f*)(orow + 16) = t;                         // cols 16..31
#pragma unroll
        for (int r = 0; r < 8; ++r) t[r] = ot2[r] * inv;
        *(v8f*)(orow + 32) = t;                         // cols 32..47
#pragma unroll
        for (int r = 0; r < 8; ++r) t[r] = ot3[r] * inv;
        *(v8f*)(orow + 48) = t;                         // cols 48..63
    }
}

// ---------------------------------------------------------------------------
extern "C" void kernel_launch(void* const* d_in, const int* in_sizes, int n_in,
                              void* d_out, int out_size, void* d_ws, size_t ws_size,
                              hipStream_t stream) {
    const float* q  = (const float*)d_in[0];
    const float* k  = (const float*)d_in[1];
    const float* v  = (const float*)d_in[2];
    const float* Wq = (const float*)d_in[3];
    const float* bq = (const float*)d_in[4];
    const float* Wk = (const float*)d_in[5];
    const float* bk = (const float*)d_in[6];
    const float* Wv = (const float*)d_in[7];
    const float* bv = (const float*)d_in[8];
    float* out = (float*)d_out;

    unsigned short* Qp = (unsigned short*)d_ws;                 // 4096*64 bf16
    unsigned short* Kp = Qp + (size_t)NQ_ * AD_;                // 8192*64 bf16
    unsigned short* Vp = Kp + (size_t)NK_ * AD_;                // 8192*64 bf16

    // 1/sqrt(ATTN_SIZE) folded into the Q projection
    proj_kernel<<<NQ_, 64, 0, stream>>>(q, Wq, bq, Qp, 128, 0.125f);
    proj_kernel<<<NK_, 64, 0, stream>>>(k, Wk, bk, Kp, 128, 1.0f);
    proj_kernel<<<NK_, 64, 0, stream>>>(v, Wv, bv, Vp, 64, 1.0f);

    flash_attn_kernel<<<NQ_ / 128, 256, 0, stream>>>(Qp, Kp, Vp, out);
}
